// SparseConvNet_13967233646793
// MI455X (gfx1250) — matrix operand order repaired and checked by reading, not compiled, not run on hardware
//
#include <hip/hip_runtime.h>
#include <hip/hip_bf16.h>

typedef __attribute__((ext_vector_type(16))) _Float16 v16h;
typedef __attribute__((ext_vector_type(8)))  _Float16 v8h;
typedef __attribute__((ext_vector_type(8)))  float    v8f;

// ---------------------------------------------------------------------------
// mask0 = float(occ)
__global__ void k_init_mask0(const int* __restrict__ occ, float* __restrict__ m0, int n) {
    int i = blockIdx.x * blockDim.x + threadIdx.x;
    if (i < n) m0[i] = (float)occ[i];
}

// act0[v*32 + c] = f16(x[c*V + v] * m0[v])   (NCDHW f32 -> NDHWC f16, masked)
__global__ void k_init_act(const float* __restrict__ x, const float* __restrict__ m0,
                           _Float16* __restrict__ act, int total, int V) {
    int i = blockIdx.x * blockDim.x + threadIdx.x;
    if (i >= total) return;
    int v = i >> 5;
    int c = i & 31;
    act[i] = (_Float16)(x[c * V + v] * m0[v]);
}

// 3x3x3 max-pool stride 2 pad 1 on mask, binarized
__global__ void k_pool_mask(const float* __restrict__ mi, float* __restrict__ mo,
                            int Din, int Dout) {
    int n = blockIdx.x * blockDim.x + threadIdx.x;
    int N = Dout * Dout * Dout;
    if (n >= N) return;
    int xo = n % Dout, yo = (n / Dout) % Dout, zo = n / (Dout * Dout);
    float mx = 0.f;
    for (int dz = 0; dz < 3; ++dz)
        for (int dy = 0; dy < 3; ++dy)
            for (int dx = 0; dx < 3; ++dx) {
                int z = zo * 2 - 1 + dz, y = yo * 2 - 1 + dy, x = xo * 2 - 1 + dx;
                if ((unsigned)z < (unsigned)Din && (unsigned)y < (unsigned)Din &&
                    (unsigned)x < (unsigned)Din)
                    mx = fmaxf(mx, mi[(z * Din + y) * Din + x]);
            }
    mo[n] = (mx > 0.f) ? 1.f : 0.f;
}

// ---------------------------------------------------------------------------
// Pack f32 weights (cout,cin,3,3,3) into WMMA A-fragment lane order, f16.
// Chunk q = tap*cbc + cb; within chunk: 32 lanes x 16 halves,
//   M = lane&15 ; K = h + ((lane>>4)<<4)
__global__ void k_pack_w(const float* __restrict__ w, _Float16* __restrict__ wpk,
                         int Cin, int Cout) {
    int i = blockIdx.x * blockDim.x + threadIdx.x;
    int total = Cout * Cin * 27;
    if (i >= total) return;
    int h    = i & 15;
    int lane = (i >> 4) & 31;
    int cbc  = Cin >> 5;
    int q    = (i >> 9) % (27 * cbc);
    int mt   = (i >> 9) / (27 * cbc);
    int t    = q / cbc;
    int cb   = q % cbc;
    int k    = h + ((lane >> 4) << 4);
    int co   = mt * 16 + (lane & 15);
    int ci   = cb * 32 + k;
    wpk[i] = (_Float16)w[(co * Cin + ci) * 27 + t];
}

// ---------------------------------------------------------------------------
// Implicit-GEMM 3x3x3 conv (pad 1) via v_wmma_f32_16x16x32_f16.
// One wave computes a 32(cout) x 32(voxel) tile: 2x2 register blocking,
// 4 WMMAs per 2 A-frag + 2 B-frag loads. Fused BN affine + ReLU + mask.
// act layouts are voxel-major f16 [D^3][C]; optional f32 NCDHW tap output.
template <int CBC, int STRIDE>
__global__ void __launch_bounds__(256) k_conv_wmma(
    const _Float16* __restrict__ act_in, _Float16* __restrict__ act_out,
    float* __restrict__ out_f32, const _Float16* __restrict__ wpk,
    const float* __restrict__ bnb, const float* __restrict__ bng,
    const float* __restrict__ bnm, const float* __restrict__ bnv,
    const float* __restrict__ mask,
    int Cout, int Din, int Dout, int ldout, int lntn, int totalTiles) {
    int wave = (blockIdx.x * blockDim.x + threadIdx.x) >> 5;
    if (wave >= totalTiles) return;
    int lane = threadIdx.x & 31;
    int nt = wave & ((1 << lntn) - 1);
    int mt = wave >> lntn;

    int n0 = nt * 32 + (lane & 15);              // first voxel column
    int n1 = n0 + 16;                            // second voxel column
    int dm = Dout - 1;
    int x0 = (n0 & dm) * STRIDE - 1;
    int y0 = ((n0 >> ldout) & dm) * STRIDE - 1;
    int z0 = (n0 >> (2 * ldout)) * STRIDE - 1;
    int x1 = (n1 & dm) * STRIDE - 1;
    int y1 = ((n1 >> ldout) & dm) * STRIDE - 1;
    int z1 = (n1 >> (2 * ldout)) * STRIDE - 1;

    const int Cin = CBC * 32;
    int koff = (lane >> 4) << 4;                 // B: upper lanes hold K=16..31
    const _Float16* wb0 = wpk + (size_t)(2 * mt) * 27 * CBC * 512 + lane * 16;
    const _Float16* wb1 = wb0 + (size_t)27 * CBC * 512;

    v8f acc00 = {}, acc01 = {}, acc10 = {}, acc11 = {};
    for (int t = 0; t < 27; ++t) {
        int dz = t / 9, dy = (t / 3) % 3, dx = t % 3;
        int za = z0 + dz, ya = y0 + dy, xa = x0 + dx;
        int zb = z1 + dz, yb = y1 + dy, xb = x1 + dx;
        bool inb0 = (unsigned)za < (unsigned)Din && (unsigned)ya < (unsigned)Din &&
                    (unsigned)xa < (unsigned)Din;
        bool inb1 = (unsigned)zb < (unsigned)Din && (unsigned)yb < (unsigned)Din &&
                    (unsigned)xb < (unsigned)Din;
        const _Float16* p0 = act_in + (size_t)((za * Din + ya) * Din + xa) * Cin + koff;
        const _Float16* p1 = act_in + (size_t)((zb * Din + yb) * Din + xb) * Cin + koff;
        const _Float16* wt0 = wb0 + (size_t)t * CBC * 512;
        const _Float16* wt1 = wb1 + (size_t)t * CBC * 512;
        #pragma unroll
        for (int cb = 0; cb < CBC; ++cb) {
            v16h a0 = *(const v16h*)(wt0 + cb * 512);
            v16h a1 = *(const v16h*)(wt1 + cb * 512);
            v16h b0 = {}, b1 = {};
            if (inb0) b0 = *(const v16h*)(p0 + cb * 32);
            if (inb1) b1 = *(const v16h*)(p1 + cb * 32);
            acc00 = __builtin_amdgcn_wmma_f32_16x16x32_f16(false, a0, false, b0,
                                                           (short)0, acc00, false, false);
            acc01 = __builtin_amdgcn_wmma_f32_16x16x32_f16(false, a0, false, b1,
                                                           (short)0, acc01, false, false);
            acc10 = __builtin_amdgcn_wmma_f32_16x16x32_f16(false, a1, false, b0,
                                                           (short)0, acc10, false, false);
            acc11 = __builtin_amdgcn_wmma_f32_16x16x32_f16(false, a1, false, b1,
                                                           (short)0, acc11, false, false);
        }
    }

    // Epilogue: lane holds D(M = r + 8*(lane>>4), N = lane&15) per sub-tile.
    const v8f* accp[2][2] = {{&acc00, &acc01}, {&acc10, &acc11}};
    float mk[2] = {mask[n0], mask[n1]};
    int nn[2] = {n0, n1};
    int mhi = (lane >> 4) << 3;
    int Nvox = 1 << (3 * ldout);
    #pragma unroll
    for (int mi = 0; mi < 2; ++mi) {
        int cobase = mt * 32 + mi * 16 + mhi;
        float sc[8], bi[8];
        #pragma unroll
        for (int r = 0; r < 8; ++r) {
            int c = cobase + r;
            sc[r] = bng[c] / sqrtf(bnv[c] + 1e-3f);
            bi[r] = bnb[c] - bnm[c] * sc[r];
        }
        #pragma unroll
        for (int ni = 0; ni < 2; ++ni) {
            const v8f& a = *accp[mi][ni];
            v8h outh;
            float vals[8];
            #pragma unroll
            for (int r = 0; r < 8; ++r) {
                float vv = fmaxf(a[r] * sc[r] + bi[r], 0.f) * mk[ni];
                vals[r] = vv;
                outh[r] = (_Float16)vv;
            }
            *(v8h*)(act_out + (size_t)nn[ni] * Cout + cobase) = outh;  // 16B store
            if (out_f32) {
                #pragma unroll
                for (int r = 0; r < 8; ++r)
                    out_f32[(size_t)(cobase + r) * Nvox + nn[ni]] = vals[r];
            }
        }
    }
}

// ---------------------------------------------------------------------------
extern "C" void kernel_launch(void* const* d_in, const int* in_sizes, int n_in,
                              void* d_out, int out_size, void* d_ws, size_t ws_size,
                              hipStream_t stream) {
    (void)in_sizes; (void)n_in; (void)out_size; (void)ws_size;
    const float* x  = (const float*)d_in[0];
    const int*  occ = (const int*)d_in[1];
    float* out = (float*)d_out;

    char* ws = (char*)d_ws;
    _Float16* actA = (_Float16*)ws;                                 // 16 MB
    _Float16* actB = (_Float16*)(ws + (size_t)16 * 1024 * 1024);    // 16 MB
    _Float16* wpk  = (_Float16*)(ws + (size_t)32 * 1024 * 1024);    // <= 884,736 B
    float* m0 = (float*)(ws + (size_t)33 * 1024 * 1024);            // masks
    float* m1 = m0 + 262144;
    float* m2 = m1 + 32768;
    float* m3 = m2 + 4096;
    float* m4 = m3 + 512;

    k_init_mask0<<<(262144 + 255) / 256, 256, 0, stream>>>(occ, m0, 262144);
    k_init_act<<<(262144 * 32 + 255) / 256, 256, 0, stream>>>(x, m0, actA,
                                                              262144 * 32, 262144);

    // One conv layer: pack weights then implicit-GEMM WMMA conv (templated).
    auto run_layer = [&](int pbase, const _Float16* ain, _Float16* aout, float* outF,
                         const float* mask, int Cin, int Cout, int Din, int Dout,
                         int ldo, int stride) {
        const float* wsrc = (const float*)d_in[pbase + 4];
        int totW = Cout * Cin * 27;
        k_pack_w<<<(totW + 255) / 256, 256, 0, stream>>>(wsrc, wpk, Cin, Cout);
        int lntn = 3 * ldo - 5;                       // log2(Nvox/32)
        int totalTiles = (1 << lntn) * (Cout / 32);
        int blocks = (totalTiles * 32 + 255) / 256;
        const float* b = (const float*)d_in[pbase + 0];
        const float* g = (const float*)d_in[pbase + 1];
        const float* m = (const float*)d_in[pbase + 2];
        const float* v = (const float*)d_in[pbase + 3];
        int cbc = Cin >> 5;
        #define LAUNCH(CBCv, Sv)                                                   \
            k_conv_wmma<CBCv, Sv><<<blocks, 256, 0, stream>>>(                     \
                ain, aout, outF, wpk, b, g, m, v, mask, Cout, Din, Dout, ldo,      \
                lntn, totalTiles)
        if (stride == 1) {
            if (cbc == 1)      LAUNCH(1, 1);
            else if (cbc == 2) LAUNCH(2, 1);
            else               LAUNCH(4, 1);
        } else {
            if (cbc == 1)      LAUNCH(1, 2);
            else if (cbc == 2) LAUNCH(2, 2);
            else               LAUNCH(4, 2);
        }
        #undef LAUNCH
    };

    // Param flat order (pytree: dict keys alphabetical, per-layer b,g,m,v,w):
    // conv0:2..11  conv1:12..21  conv2:22..36  conv3:37..51  conv4:52..66
    // down0:67  down1:72  down2:77  down3:82
    run_layer( 2, actA, actB, nullptr, m0, 32, 32, 64, 64, 6, 1);   // conv0[0]
    run_layer( 7, actB, actA, nullptr, m0, 32, 32, 64, 64, 6, 1);   // conv0[1]
    k_pool_mask<<<(32768 + 255) / 256, 256, 0, stream>>>(m0, m1, 64, 32);
    run_layer(67, actA, actB, nullptr, m1, 32, 32, 64, 32, 5, 2);   // down0
    run_layer(12, actB, actA, nullptr, m1, 32, 32, 32, 32, 5, 1);   // conv1[0]
    run_layer(17, actA, actB, out + 0, m1, 32, 32, 32, 32, 5, 1);   // conv1[1] -> net1
    k_pool_mask<<<(4096 + 255) / 256, 256, 0, stream>>>(m1, m2, 32, 16);
    run_layer(72, actB, actA, nullptr, m2, 32, 64, 32, 16, 4, 2);   // down1
    run_layer(22, actA, actB, nullptr, m2, 64, 64, 16, 16, 4, 1);   // conv2[0]
    run_layer(27, actB, actA, nullptr, m2, 64, 64, 16, 16, 4, 1);   // conv2[1]
    run_layer(32, actA, actB, out + 1048576, m2, 64, 64, 16, 16, 4, 1); // -> net2
    k_pool_mask<<<(512 + 255) / 256, 256, 0, stream>>>(m2, m3, 16, 8);
    run_layer(77, actB, actA, nullptr, m3, 64, 128, 16, 8, 3, 2);   // down2
    run_layer(37, actA, actB, nullptr, m3, 128, 128, 8, 8, 3, 1);   // conv3[0]
    run_layer(42, actB, actA, nullptr, m3, 128, 128, 8, 8, 3, 1);   // conv3[1]
    run_layer(47, actA, actB, out + 1310720, m3, 128, 128, 8, 8, 3, 1); // -> net3
    k_pool_mask<<<1, 256, 0, stream>>>(m3, m4, 8, 4);
    run_layer(82, actB, actA, nullptr, m4, 128, 128, 8, 4, 2, 2);   // down3
    run_layer(52, actA, actB, nullptr, m4, 128, 128, 4, 4, 2, 1);   // conv4[0]
    run_layer(57, actB, actA, nullptr, m4, 128, 128, 4, 4, 2, 1);   // conv4[1]
    run_layer(62, actA, actB, out + 1376256, m4, 128, 128, 4, 4, 2, 1); // -> net4
}
